// CPI_DGLLife_3255585210691
// MI455X (gfx1250) — matrix operand order repaired and checked by reading, not compiled, not run on hardware
//
#include <hip/hip_runtime.h>
#include <hip/hip_bf16.h>

#define NNODES 65536
#define NEDGES 262144
#define BATCH  512
#define LSEQ   1000
#define INDIM  74
#define HID    128
#define VOCAB  25

typedef __attribute__((ext_vector_type(16))) _Float16 v16h;
typedef __attribute__((ext_vector_type(8)))  _Float16 v8h;
typedef __attribute__((ext_vector_type(8)))  float    v8f;

// LDS row strides (halves), padded so 16-lane b128 A-loads are bank-conflict free
#define LDA_CONV 136   // 128 cols + 8 pad  (68 DW stride -> banks 4m)
#define LDA_GC   104   // 96 cols + 8 pad   (52 DW stride -> banks 4*(13m%16))
#define LDA_HEAD 264   // 256 cols + 8 pad  (132 DW stride -> banks 4m)

// ---------------- WMMA helpers (gfx1250 wave32, 16x16x32 f16->f32) -----------

__device__ inline v8f wmma16(v16h a, v16h b, v8f c) {
  return __builtin_amdgcn_wmma_f32_16x16x32_f16(
      /*neg_a=*/false, a, /*neg_b=*/false, b,
      /*c_mod=*/(short)0, c, /*reuse_a=*/false, /*reuse_b=*/false);
}

// A tile 16(M)x32(K) from row-major f16, base = &A[row0][k0], stride lda.
// Per-lane: two contiguous 16B chunks -> 2x ds/global_load_b128.
__device__ inline v16h load_a(const _Float16* A, int lda, int lane) {
  int m = lane & 15, hi = lane >> 4;
  const _Float16* r = A + m * lda + hi * 16;
  v16h a;
#pragma unroll
  for (int j = 0; j < 8; ++j) a[j] = r[j];
#pragma unroll
  for (int j = 0; j < 8; ++j) a[8 + j] = r[8 + j];
  return a;
}

// B fragment from packed fragment-major buffer: frag blocks of 512 halves,
// lane's 16 halves contiguous -> 2x global_load_b128, no packing VALU.
__device__ inline v16h load_bp(const _Float16* Bp, int frag, int lane) {
  const _Float16* p = Bp + (size_t)frag * 512 + lane * 16;
  v16h b;
#pragma unroll
  for (int j = 0; j < 16; ++j) b[j] = p[j];
  return b;
}

// ---------------- graph-side kernels -----------------------------------------

__global__ void k_init(float* deg_out, float* deg_in, float* hg, float* pmax) {
  int i = blockIdx.x * 256 + threadIdx.x;           // 65536 == N == BATCH*HID
  if (i < NNODES) { deg_out[i] = 1.f; deg_in[i] = 1.f; }
  if (i < BATCH * HID) { hg[i] = 0.f; pmax[i] = 0.f; }
}

__global__ void k_deg(const int* __restrict__ src, const int* __restrict__ dst,
                      float* dOut, float* dIn) {
  int e = blockIdx.x * 256 + threadIdx.x;
  if (e < NEDGES) { atomicAdd(dOut + src[e], 1.f); atomicAdd(dIn + dst[e], 1.f); }
}

__global__ void k_scale(const float* __restrict__ nf, const float* __restrict__ dOut,
                        float* tmp, float* accum) {
  int i = blockIdx.x * 256 + threadIdx.x;
  if (i < NNODES * INDIM) {
    int n = i / INDIM;
    float v = nf[i] * rsqrtf(dOut[n]);
    tmp[i] = v; accum[i] = v;                       // accum starts with self-loop term
  }
}

__global__ void k_scatter(const int* __restrict__ src, const int* __restrict__ dst,
                          const float* __restrict__ tmp, float* accum) {
  int e = blockIdx.x;
  int c = threadIdx.x;
  if (c < INDIM) {
    int s = src[e], d = dst[e];
    atomicAdd(&accum[d * INDIM + c], tmp[s * INDIM + c]);
  }
}

// ---------------- weight prep: pack into WMMA fragment-major layout ----------
// frag = (k0/32)*(Np/16) + (n0/16); element (lane,e): K = k0+(lane>>4)*16+e,
// N = n0+(lane&15). Zero-padded beyond (R,C).

__global__ void k_pack_b(const float* __restrict__ S, _Float16* D, int R, int C, int Kp, int Np) {
  int i = blockIdx.x * 256 + threadIdx.x;           // over Kp*Np
  if (i >= Kp * Np) return;
  int frag = i >> 9, rem = i & 511;
  int lane = rem >> 4, e = rem & 15;
  int ntiles = Np >> 4;
  int k = (frag / ntiles) * 32 + (lane >> 4) * 16 + e;
  int n = (frag % ntiles) * 16 + (lane & 15);
  float v = (k < R && n < C) ? S[k * C + n] : 0.f;
  D[i] = (_Float16)v;
}

// conv kernel K[Cout][Cin][3] viewed as B matrix [3*CinP][CoutP], k = dk*CinP+ci
__global__ void k_pack_conv(const float* __restrict__ Ksrc, _Float16* D,
                            int Cin, int Cout, int CinP, int CoutP) {
  int tot = 3 * CinP * CoutP;
  int i = blockIdx.x * 256 + threadIdx.x;
  if (i >= tot) return;
  int frag = i >> 9, rem = i & 511;
  int lane = rem >> 4, e = rem & 15;
  int ntiles = CoutP >> 4;
  int k = (frag / ntiles) * 32 + (lane >> 4) * 16 + e;
  int n = (frag % ntiles) * 16 + (lane & 15);
  int dk = k / CinP, ci = k % CinP;
  float v = (ci < Cin && n < Cout) ? Ksrc[(n * Cin + ci) * 3 + dk] : 0.f;
  D[i] = (_Float16)v;
}

__global__ void k_pad_bias(const float* __restrict__ s, float* d, int C, int Cp) {
  int i = blockIdx.x * 256 + threadIdx.x;
  if (i < Cp) d[i] = (i < C) ? s[i] : 0.f;
}

__global__ void k_cvt_f16(const float* __restrict__ S, _Float16* D, int n) {
  int i = blockIdx.x * 256 + threadIdx.x;
  if (i < n) D[i] = (_Float16)S[i];
}

// fuse readout linears: Wc = Win@Wout written packed; bc = bin@Wout + bout
__global__ void k_fuse_ro(const float* __restrict__ Win, const float* __restrict__ bin,
                          const float* __restrict__ Wout, const float* __restrict__ bout,
                          _Float16* WcP, float* bc) {
  int i = blockIdx.x, j = threadIdx.x;              // Wc[i][j], K=i, N=j
  float s = 0.f;
  for (int k = 0; k < HID; ++k) s += Win[i * HID + k] * Wout[k * HID + j];
  int frag = (i >> 5) * 8 + (j >> 4);
  int lane = (j & 15) + (((i >> 4) & 1) << 4);
  int e = i & 15;
  WcP[frag * 512 + lane * 16 + e] = (_Float16)s;
  if (i == 0) {
    float t = bout[j];
    for (int k = 0; k < HID; ++k) t += bin[k] * Wout[k * HID + j];
    bc[j] = t;
  }
}

// ---------------- GraphConv GEMM: h = relu((accum * rsqrt(deg_in)) @ Wgc + b) -

__global__ void k_gc(const float* __restrict__ accum, const float* __restrict__ dIn,
                     const _Float16* __restrict__ WgcP, const float* __restrict__ bgc,
                     _Float16* h) {
  __shared__ _Float16 As[128][LDA_GC];
  int row0 = blockIdx.x * 128;
  int tid = threadIdx.x;
  for (int i = tid; i < 128 * 96; i += 256) {
    int r = i / 96, c = i % 96;
    int n = row0 + r;
    float v = 0.f;
    if (c < INDIM) v = accum[n * INDIM + c] * rsqrtf(dIn[n]);
    As[r][c] = (_Float16)v;
  }
  __syncthreads();
  int wave = tid >> 5, lane = tid & 31, n = lane & 15, hi = lane >> 4;
  int nt = wave;                                    // 8 waves == 8 N-tiles
  v8f acc[8] = {};
  for (int ks = 0; ks < 3; ++ks) {                  // K = 96
    v16h bfrag = load_bp(WgcP, ks * 8 + nt, lane);
#pragma unroll
    for (int mt = 0; mt < 8; ++mt)
      acc[mt] = wmma16(load_a(&As[mt * 16][ks * 32], LDA_GC, lane), bfrag, acc[mt]);
  }
  float bv = bgc[nt * 16 + n];
#pragma unroll
  for (int mt = 0; mt < 8; ++mt)
#pragma unroll
    for (int r = 0; r < 8; ++r) {
      float v = acc[mt][r] + bv; v = v > 0.f ? v : 0.f;
      h[(row0 + mt * 16 + r + 8 * hi) * 128 + nt * 16 + n] = (_Float16)v;
    }
}

// ------------- fused readout GEMM + segment-sum into hg ----------------------

__global__ void k_ro(const _Float16* __restrict__ h, const _Float16* __restrict__ WcP,
                     const float* __restrict__ bc, const int* __restrict__ gids,
                     float* hg) {
  int row0 = blockIdx.x * 128;
  int tid = threadIdx.x;
  int wave = tid >> 5, lane = tid & 31, n = lane & 15, hi = lane >> 4;
  int nt = wave;
  v8f acc[8] = {};
  for (int ks = 0; ks < 4; ++ks) {                  // K = 128
    v16h bfrag = load_bp(WcP, ks * 8 + nt, lane);
#pragma unroll
    for (int mt = 0; mt < 8; ++mt)
      acc[mt] = wmma16(load_a(h + (row0 + mt * 16) * 128 + ks * 32, 128, lane),
                       bfrag, acc[mt]);
  }
  float bv = bc[nt * 16 + n];
#pragma unroll
  for (int mt = 0; mt < 8; ++mt)
#pragma unroll
    for (int r = 0; r < 8; ++r) {
      int row = row0 + mt * 16 + r + 8 * hi;
      atomicAdd(&hg[gids[row] * HID + nt * 16 + n], acc[mt][r] + bv);
    }
}

// ------------- compound FC stack: cv = relu(relu(relu(hg)@W1+b1)@W2+b2) ------

__global__ void k_cfc(const float* __restrict__ hg, const _Float16* __restrict__ W1,
                      const float* __restrict__ b1, const _Float16* __restrict__ W2,
                      const float* __restrict__ b2, _Float16* cv) {
  __shared__ _Float16 Z[64][LDA_CONV];
  __shared__ _Float16 Z2[64][LDA_CONV];
  int row0 = blockIdx.x * 64;
  int tid = threadIdx.x;
  for (int i = tid; i < 64 * 128; i += 256) {
    int r = i >> 7, c = i & 127;
    float v = hg[(row0 + r) * HID + c];
    Z[r][c] = (_Float16)(v > 0.f ? v : 0.f);
  }
  __syncthreads();
  int wave = tid >> 5, lane = tid & 31, n = lane & 15, hi = lane >> 4;
  int nt = wave;
  {
    v8f acc[4] = {};
    for (int ks = 0; ks < 4; ++ks) {
      v16h bfrag = load_bp(W1, ks * 8 + nt, lane);
#pragma unroll
      for (int mt = 0; mt < 4; ++mt)
        acc[mt] = wmma16(load_a(&Z[mt * 16][ks * 32], LDA_CONV, lane), bfrag, acc[mt]);
    }
    float bv = b1[nt * 16 + n];
#pragma unroll
    for (int mt = 0; mt < 4; ++mt)
#pragma unroll
      for (int r = 0; r < 8; ++r) {
        float v = acc[mt][r] + bv;
        Z2[mt * 16 + r + 8 * hi][nt * 16 + n] = (_Float16)(v > 0.f ? v : 0.f);
      }
  }
  __syncthreads();
  {
    v8f acc[4] = {};
    for (int ks = 0; ks < 4; ++ks) {
      v16h bfrag = load_bp(W2, ks * 8 + nt, lane);
#pragma unroll
      for (int mt = 0; mt < 4; ++mt)
        acc[mt] = wmma16(load_a(&Z2[mt * 16][ks * 32], LDA_CONV, lane), bfrag, acc[mt]);
    }
    float bv = b2[nt * 16 + n];
#pragma unroll
    for (int mt = 0; mt < 4; ++mt)
#pragma unroll
      for (int r = 0; r < 8; ++r) {
        float v = acc[mt][r] + bv;
        cv[(row0 + mt * 16 + r + 8 * hi) * 128 + nt * 16 + n] = (_Float16)(v > 0.f ? v : 0.f);
      }
  }
}

// ------------- protein branch: embed + 4 fused conv layers + maxpool ---------
// LDS stage buffers: phys row = p+1, p in [-1, 80]; loaded valid p in [0,72).
// After 4 layers valid output region is p in [4,68) == 64 positions.

__device__ inline void conv_layer(_Float16 (*Sin)[LDA_CONV], _Float16 (*Sout)[LDA_CONV],
                                  const _Float16* __restrict__ W,
                                  const float* __restrict__ bias,
                                  int CinP, int CoutP, int l0, int tid) {
  int wave = tid >> 5, lane = tid & 31, n = lane & 15, hi = lane >> 4;
  int ntiles = CoutP >> 4;
  int kc_per_dk = CinP >> 5;
  for (int nt = wave; nt < ntiles; nt += 8) {
    v8f acc[5] = {};                                 // 5 mt tiles, 80 positions
    for (int dk = 0; dk < 3; ++dk) {
      for (int kc = 0; kc < kc_per_dk; ++kc) {
        v16h bfrag = load_bp(W, (dk * kc_per_dk + kc) * ntiles + nt, lane);
#pragma unroll
        for (int mt = 0; mt < 5; ++mt) {
          // output p = mt*16+m reads input phys row p+dk (= p-1+dk, +1 offset)
          v16h afrag = load_a(&Sin[mt * 16 + dk][kc * 32], LDA_CONV, lane);
          acc[mt] = wmma16(afrag, bfrag, acc[mt]);
        }
      }
    }
    float bv = bias[nt * 16 + n];
#pragma unroll
    for (int mt = 0; mt < 5; ++mt)
#pragma unroll
      for (int r = 0; r < 8; ++r) {
        int p = mt * 16 + r + 8 * hi;
        int l = l0 - 4 + p;
        float v = acc[mt][r] + bv; v = v > 0.f ? v : 0.f;
        if (l < 0 || l >= LSEQ) v = 0.f;             // emulate zero padding
        Sout[p + 1][nt * 16 + n] = (_Float16)v;
      }
  }
}

__global__ void k_prot(const int* __restrict__ seq, const _Float16* __restrict__ embedH,
                       const _Float16* __restrict__ W1, const float* __restrict__ cb1,
                       const _Float16* __restrict__ W2, const float* __restrict__ cb2,
                       const _Float16* __restrict__ W3, const float* __restrict__ cb3p,
                       const _Float16* __restrict__ W4, const float* __restrict__ cb4,
                       float* pmax) {
  __shared__ _Float16 S0[82][LDA_CONV];
  __shared__ _Float16 S1[82][LDA_CONV];
  int b = blockIdx.y;
  int l0 = blockIdx.x * 64;
  int tid = threadIdx.x;

  // stage 0: f16 embedding gather in 16B chunks into p in [0,72); zero halos
  for (int i = tid; i < 82 * 16; i += 256) {
    int pr = i >> 4, ch = i & 15;
    int p = pr - 1;
    v8h v = {};
    if (p >= 0 && p < 72) {
      int l = l0 - 4 + p;
      if (l >= 0 && l < LSEQ)
        v = *(const v8h*)(embedH + seq[b * LSEQ + l] * HID + ch * 8);
    }
    *(v8h*)(&S0[pr][ch * 8]) = v;
  }
  for (int i = tid; i < 128; i += 256) { S1[0][i] = (_Float16)0.f; S1[81][i] = (_Float16)0.f; }
  __syncthreads();

  conv_layer(S0, S1, W1, cb1, 128, 96, l0, tid);  __syncthreads();  // 128 -> 96
  conv_layer(S1, S0, W2, cb2, 96, 128, l0, tid);  __syncthreads();  // 96  -> 128
  conv_layer(S0, S1, W3, cb3p, 128, 96, l0, tid); __syncthreads();  // 128 -> 74(pad 96)
  conv_layer(S1, S0, W4, cb4, 96, 128, l0, tid);  __syncthreads();  // 74(pad) -> 128

  // max over valid p in [4,68); post-ReLU values >= 0 -> uint atomicMax valid
  int c = tid & 127, half = tid >> 7;
  float m = 0.f;
  for (int p = 4 + half; p < 68; p += 2) {
    int l = l0 - 4 + p;
    if (l < LSEQ) { float v = (float)S0[p + 1][c]; m = m > v ? m : v; }
  }
  atomicMax((unsigned int*)&pmax[b * HID + c], __float_as_uint(m));
}

// ------------- CPI head ------------------------------------------------------

__global__ void k_head(const _Float16* __restrict__ cv, const float* __restrict__ pmax,
                       const _Float16* __restrict__ Wf1, const float* __restrict__ bf1,
                       const float* __restrict__ Wf2, const float* __restrict__ bf2,
                       float* out) {
  __shared__ _Float16 Z[64][LDA_HEAD];
  __shared__ _Float16 Z2[64][LDA_HEAD];
  int row0 = blockIdx.x * 64;
  int tid = threadIdx.x;
  for (int i = tid; i < 64 * 256; i += 256) {
    int r = i >> 8, c = i & 255;
    float v = (c < 128) ? (float)cv[(row0 + r) * 128 + c]
                        : pmax[(row0 + r) * 128 + (c - 128)];
    Z[r][c] = (_Float16)v;
  }
  __syncthreads();
  int wave = tid >> 5, lane = tid & 31, n = lane & 15, hi = lane >> 4;
  for (int nt = wave; nt < 16; nt += 8) {           // Np = 256 -> 16 N-tiles
    v8f acc[4] = {};
    for (int ks = 0; ks < 8; ++ks) {                // K = 256
      v16h bfrag = load_bp(Wf1, ks * 16 + nt, lane);
#pragma unroll
      for (int mt = 0; mt < 4; ++mt)
        acc[mt] = wmma16(load_a(&Z[mt * 16][ks * 32], LDA_HEAD, lane), bfrag, acc[mt]);
    }
    float bv = bf1[nt * 16 + n];
#pragma unroll
    for (int mt = 0; mt < 4; ++mt)
#pragma unroll
      for (int r = 0; r < 8; ++r) {
        float v = acc[mt][r] + bv;
        Z2[mt * 16 + r + 8 * hi][nt * 16 + n] = (_Float16)(v > 0.f ? v : 0.f);
      }
  }
  __syncthreads();
  if (tid < 64) {
    float s = bf2[0];
    for (int k = 0; k < 256; ++k) s += (float)Z2[tid][k] * Wf2[k];
    out[row0 + tid] = 1.f / (1.f + __expf(-s));
  }
}

// ------------- launcher ------------------------------------------------------

extern "C" void kernel_launch(void* const* d_in, const int* in_sizes, int n_in,
                              void* d_out, int out_size, void* d_ws, size_t ws_size,
                              hipStream_t stream) {
  const float* node_feats = (const float*)d_in[0];
  const float* W_gc   = (const float*)d_in[1];
  const float* b_gc   = (const float*)d_in[2];
  const float* W_ro_in  = (const float*)d_in[3];
  const float* b_ro_in  = (const float*)d_in[4];
  const float* W_ro_out = (const float*)d_in[5];
  const float* b_ro_out = (const float*)d_in[6];
  const float* Wc1 = (const float*)d_in[7];  const float* bc1 = (const float*)d_in[8];
  const float* Wc2 = (const float*)d_in[9];  const float* bc2 = (const float*)d_in[10];
  const float* embed = (const float*)d_in[11];
  const float* K1 = (const float*)d_in[12]; const float* cb1 = (const float*)d_in[13];
  const float* K2 = (const float*)d_in[14]; const float* cb2 = (const float*)d_in[15];
  const float* K3 = (const float*)d_in[16]; const float* cb3 = (const float*)d_in[17];
  const float* K4 = (const float*)d_in[18]; const float* cb4 = (const float*)d_in[19];
  const float* Wf1 = (const float*)d_in[20]; const float* bf1 = (const float*)d_in[21];
  const float* Wf2 = (const float*)d_in[22]; const float* bf2 = (const float*)d_in[23];
  const int* edge_src = (const int*)d_in[24];
  const int* edge_dst = (const int*)d_in[25];
  const int* graph_ids = (const int*)d_in[26];
  const int* protein_seq = (const int*)d_in[27];
  float* out = (float*)d_out;

  // bump allocator over d_ws
  char* w = (char*)d_ws;
  auto alloc = [&](size_t bytes) -> void* {
    void* p = (void*)w; w += (bytes + 255) & ~(size_t)255; return p;
  };
  float* deg_out = (float*)alloc(NNODES * 4);
  float* deg_in  = (float*)alloc(NNODES * 4);
  float* tmp     = (float*)alloc((size_t)NNODES * INDIM * 4);
  float* accum   = (float*)alloc((size_t)NNODES * INDIM * 4);
  _Float16* h    = (_Float16*)alloc((size_t)NNODES * HID * 2);
  _Float16* WgcP = (_Float16*)alloc(96 * 128 * 2);
  _Float16* WcP  = (_Float16*)alloc(128 * 128 * 2);
  float* bc      = (float*)alloc(128 * 4);
  _Float16* Wc1P = (_Float16*)alloc(128 * 128 * 2);
  _Float16* Wc2P = (_Float16*)alloc(128 * 128 * 2);
  _Float16* Wf1P = (_Float16*)alloc(256 * 256 * 2);
  _Float16* Wcv1 = (_Float16*)alloc(384 * 96 * 2);   // 128p -> 96p
  _Float16* Wcv2 = (_Float16*)alloc(288 * 128 * 2);  // 96p  -> 128p
  _Float16* Wcv3 = (_Float16*)alloc(384 * 96 * 2);   // 128p -> 96p (74 padded)
  _Float16* Wcv4 = (_Float16*)alloc(288 * 128 * 2);  // 96p  -> 128p
  float* cb3p    = (float*)alloc(96 * 4);
  _Float16* embedH = (_Float16*)alloc(VOCAB * HID * 2);
  float* hg      = (float*)alloc(BATCH * HID * 4);
  _Float16* cv   = (_Float16*)alloc(BATCH * HID * 2);
  float* pmax    = (float*)alloc(BATCH * HID * 4);

  // graph prep
  k_init<<<NNODES / 256, 256, 0, stream>>>(deg_out, deg_in, hg, pmax);
  k_deg<<<NEDGES / 256, 256, 0, stream>>>(edge_src, edge_dst, deg_out, deg_in);
  k_scale<<<(NNODES * INDIM + 255) / 256, 256, 0, stream>>>(node_feats, deg_out, tmp, accum);
  k_scatter<<<NEDGES, 128, 0, stream>>>(edge_src, edge_dst, tmp, accum);

  // weight prep (tiny, L2-resident afterwards), packed fragment-major
  k_pack_b<<<(96 * 128 + 255) / 256, 256, 0, stream>>>(W_gc, WgcP, 74, 128, 96, 128);
  k_fuse_ro<<<128, 128, 0, stream>>>(W_ro_in, b_ro_in, W_ro_out, b_ro_out, WcP, bc);
  k_pack_b<<<(128 * 128 + 255) / 256, 256, 0, stream>>>(Wc1, Wc1P, 128, 128, 128, 128);
  k_pack_b<<<(128 * 128 + 255) / 256, 256, 0, stream>>>(Wc2, Wc2P, 128, 128, 128, 128);
  k_pack_b<<<(256 * 256 + 255) / 256, 256, 0, stream>>>(Wf1, Wf1P, 256, 256, 256, 256);
  k_pack_conv<<<(3 * 128 * 96 + 255) / 256, 256, 0, stream>>>(K1, Wcv1, 128, 96, 128, 96);
  k_pack_conv<<<(3 * 96 * 128 + 255) / 256, 256, 0, stream>>>(K2, Wcv2, 96, 128, 96, 128);
  k_pack_conv<<<(3 * 128 * 96 + 255) / 256, 256, 0, stream>>>(K3, Wcv3, 128, 74, 128, 96);
  k_pack_conv<<<(3 * 96 * 128 + 255) / 256, 256, 0, stream>>>(K4, Wcv4, 74, 128, 96, 128);
  k_pad_bias<<<1, 256, 0, stream>>>(cb3, cb3p, 74, 96);
  k_cvt_f16<<<(VOCAB * HID + 255) / 256, 256, 0, stream>>>(embed, embedH, VOCAB * HID);

  // GraphConv GEMM -> h (f16)
  k_gc<<<NNODES / 128, 256, 0, stream>>>(accum, deg_in, WgcP, b_gc, h);
  // fused readout GEMM + segment sum -> hg
  k_ro<<<NNODES / 128, 256, 0, stream>>>(h, WcP, bc, graph_ids, hg);
  // compound FC stack -> cv (f16)
  k_cfc<<<BATCH / 64, 256, 0, stream>>>(hg, Wc1P, bc1, Wc2P, bc2, cv);

  // protein branch: embed + 4 convs + maxpool, fully fused
  dim3 pgrid(16, BATCH);   // 16 x 64-position tiles cover L=1000 (masked)
  k_prot<<<pgrid, 256, 0, stream>>>(protein_seq, embedH,
                                    Wcv1, cb1, Wcv2, cb2, Wcv3, cb3p, Wcv4, cb4, pmax);

  // CPI head -> sigmoid output
  k_head<<<BATCH / 64, 256, 0, stream>>>(cv, pmax, Wf1P, bf1, Wf2, bf2, out);
  (void)in_sizes; (void)n_in; (void)out_size; (void)ws_size;
}
